// GPTLike_19318762897674
// MI455X (gfx1250) — matrix operand order repaired and checked by reading, not compiled, not run on hardware
//
#include <hip/hip_runtime.h>
#include <hip/hip_bf16.h>
#include <math.h>

typedef __bf16 bf16_t;
typedef __attribute__((ext_vector_type(16))) __bf16 bf16x16;
typedef __attribute__((ext_vector_type(8)))  __bf16 bf16x8;
typedef __attribute__((ext_vector_type(8)))  float  f32x8;

namespace {
constexpr int Bn = 2, Tn = 1024, Dn = 1024, Hn = 16, FFn = 4096, Ln = 8, Vn = 32000, HDn = 64;
constexpr int Mn  = Bn * Tn;   // 2048 token rows
constexpr int D3n = 3 * Dn;    // 3072
constexpr float kScale = 0.125f;  // 1/sqrt(64)
}

union FragB16 { bf16x16 v; bf16x8 h8[2]; };

// ---------------------------------------------------------------------------
// Embedding: x[b,t,:] = tok_emb[idx[b,t]] + sinusoidal PE(t)
// ---------------------------------------------------------------------------
__global__ __launch_bounds__(256) void embed_kernel(
    const int* __restrict__ idx, const float* __restrict__ tok,
    float* __restrict__ x)
{
  long i = (long)blockIdx.x * blockDim.x + threadIdx.x;
  if (i >= (long)Mn * Dn) return;
  int d = (int)(i % Dn);
  long row = i / Dn;              // b*T + t
  int t = (int)(row % Tn);
  int token = idx[row];
  float div = __expf(-(float)(d & ~1) * (9.210340371976184f / (float)Dn));
  float ang = (float)t * div;
  x[i] = tok[(long)token * Dn + d] + ((d & 1) ? __cosf(ang) : __sinf(ang));
}

// ---------------------------------------------------------------------------
// Weight cast+transpose: W [layers, K, N] f32 -> Wt [layers, N, K] bf16
// (so WMMA B-fragments load contiguous K runs)
// ---------------------------------------------------------------------------
__global__ __launch_bounds__(256) void transpose_to_bf16(
    const float* __restrict__ W, bf16_t* __restrict__ Wt,
    int K, int N, int layers)
{
  long total = (long)layers * K * N;
  for (long i = (long)blockIdx.x * blockDim.x + threadIdx.x; i < total;
       i += (long)gridDim.x * blockDim.x) {
    long l = i / ((long)K * N);
    long r = i % ((long)K * N);
    int n = (int)(r / K);
    int k = (int)(r % K);
    Wt[i] = (bf16_t)W[l * (long)K * N + (long)k * N + n];
  }
}

// ---------------------------------------------------------------------------
// LayerNorm (f32 in) -> bf16 out, one block per row
// ---------------------------------------------------------------------------
__global__ __launch_bounds__(256) void layernorm_bf16(
    const float* __restrict__ x, const float* __restrict__ s,
    const float* __restrict__ b, bf16_t* __restrict__ o)
{
  const int row = blockIdx.x;
  const float* xr = x + (long)row * Dn;
  float sum = 0.f, sq = 0.f;
  for (int d = threadIdx.x; d < Dn; d += 256) {
    float v = xr[d]; sum += v; sq += v * v;
  }
  __shared__ float ssum[8], ssq[8];
#pragma unroll
  for (int m = 1; m < 32; m <<= 1) {
    sum += __shfl_xor(sum, m, 32);
    sq  += __shfl_xor(sq,  m, 32);
  }
  if ((threadIdx.x & 31) == 0) { ssum[threadIdx.x >> 5] = sum; ssq[threadIdx.x >> 5] = sq; }
  __syncthreads();
  sum = 0.f; sq = 0.f;
#pragma unroll
  for (int w = 0; w < 8; ++w) { sum += ssum[w]; sq += ssq[w]; }
  const float mean = sum / (float)Dn;
  const float var  = sq / (float)Dn - mean * mean;
  const float rstd = rsqrtf(var + 1e-5f);
  for (int d = threadIdx.x; d < Dn; d += 256)
    o[(long)row * Dn + d] = (bf16_t)((xr[d] - mean) * rstd * s[d] + b[d]);
}

// ---------------------------------------------------------------------------
// GEMM: C[M,N] = act(A[M,K] @ Bt[N,K]^T + bias) (+residual)
// One wave per 64x64 tile: 4 A-frags x 4 B-frags -> 16 WMMA per K-step,
// 16 global_load_b128 per K-step  =>  1 load per v_wmma (4x the intensity
// of a 16x16-per-wave scheme; keeps the XDL pipe fed).
// ---------------------------------------------------------------------------
template<int OUT_BF16, int GELU, int RES>
__global__ __launch_bounds__(256) void gemm_wmma_bf16(
    const bf16_t* __restrict__ A, const bf16_t* __restrict__ Bt,
    const float* __restrict__ bias, const float* __restrict__ resid,
    void* __restrict__ out, int M, int N, int K)
{
  const int wave = threadIdx.x >> 5;
  const int lane = threadIdx.x & 31;
  const int tiles_n = N >> 6;
  const long tile = (long)blockIdx.x * 8 + wave;
  if (tile >= (long)(M >> 6) * tiles_n) return;   // wave-uniform guard
  const int m0 = (int)(tile / tiles_n) << 6;
  const int n0 = (int)(tile % tiles_n) << 6;
  const int rc = lane & 15;
  const int hi = lane >> 4;

  // A-frag: lane row = rc, K run [hi*16 .. +15]
  const bf16_t* ap = A + (long)(m0 + rc) * K + hi * 16;
  // B-frag: lane col = rc, K run [hi*16 .. +15]
  const bf16_t* bp = Bt + (long)(n0 + rc) * K + hi * 16;

  f32x8 acc[4][4] = {};
  for (int kb = 0; kb < K; kb += 32) {
    FragB16 fa[4];
#pragma unroll
    for (int mi = 0; mi < 4; ++mi) {
      const bf16_t* p = ap + (long)mi * 16 * K + kb;
      fa[mi].h8[0] = *(const bf16x8*)(p);
      fa[mi].h8[1] = *(const bf16x8*)(p + 8);
    }
#pragma unroll
    for (int ni = 0; ni < 4; ++ni) {
      FragB16 fb;
      const bf16_t* p = bp + (long)ni * 16 * K + kb;
      fb.h8[0] = *(const bf16x8*)(p);
      fb.h8[1] = *(const bf16x8*)(p + 8);
#pragma unroll
      for (int mi = 0; mi < 4; ++mi)
        acc[mi][ni] = __builtin_amdgcn_wmma_f32_16x16x32_bf16(
            false, fa[mi].v, false, fb.v, (short)0, acc[mi][ni], false, false);
    }
  }

#pragma unroll
  for (int ni = 0; ni < 4; ++ni) {
    const int n = n0 + ni * 16 + rc;
    const float bv = bias ? bias[n] : 0.0f;
#pragma unroll
    for (int mi = 0; mi < 4; ++mi) {
#pragma unroll
      for (int r = 0; r < 8; ++r) {
        const long m = m0 + mi * 16 + r + hi * 8;  // C layout: vgpr r -> row
        float v = acc[mi][ni][r] + bv;
        if (GELU) v = 0.5f * v * (1.0f + erff(v * 0.7071067811865475f));
        if (RES)  v += resid[m * N + n];
        if (OUT_BF16) ((bf16_t*)out)[m * N + n] = (bf16_t)v;
        else          ((float*)out)[m * N + n] = v;
      }
    }
  }
}

// ---------------------------------------------------------------------------
// V transpose: qkv[:, 2D + h*HD + d] -> vT[(b*H+h)*HD + d, t]  (bf16)
// ---------------------------------------------------------------------------
__global__ __launch_bounds__(256) void vtranspose(
    const bf16_t* __restrict__ qkv, bf16_t* __restrict__ vT)
{
  long i = (long)blockIdx.x * blockDim.x + threadIdx.x;
  if (i >= (long)Bn * Dn * Tn) return;
  int t = (int)(i % Tn);
  long r = i / Tn;                // b*D + (h*HD + d)
  int hd = (int)(r % Dn);
  int b  = (int)(r / Dn);
  vT[i] = qkv[((long)b * Tn + t) * D3n + 2 * Dn + hd];
}

// ---------------------------------------------------------------------------
// Flash attention: one wave per (b, h, 16-row query tile), WMMA throughout.
// ---------------------------------------------------------------------------
__global__ __launch_bounds__(256) void attn_flash_wmma(
    const bf16_t* __restrict__ qkv, const bf16_t* __restrict__ vT,
    bf16_t* __restrict__ y)
{
  __shared__ bf16_t pshm[8][512];   // per-wave 16x32 P tile
  const int wave = threadIdx.x >> 5;
  const int lane = threadIdx.x & 31;
  const int wg = blockIdx.x * 8 + wave;     // grid sized exactly: B*H*(T/16) waves
  const int qt = wg & 63;                   // T/16 = 64
  const int hh = (wg >> 6) & 15;
  const int bb = wg >> 10;
  const int rc = lane & 15;
  const int hi = lane >> 4;
  const int qb = qt << 4;

  const bf16_t* qrow  = qkv + ((long)(bb * Tn + qb + rc)) * D3n + hh * HDn;
  const bf16_t* kbase = qkv + ((long)bb * Tn) * D3n + Dn + hh * HDn;
  const bf16_t* vbase = vT + ((long)(bb * Hn + hh)) * HDn * Tn;

  // Q A-fragments for d-steps 0..31 and 32..63 (loop-invariant)
  FragB16 q0, q1;
  q0.h8[0] = *(const bf16x8*)(qrow + hi * 16);
  q0.h8[1] = *(const bf16x8*)(qrow + hi * 16 + 8);
  q1.h8[0] = *(const bf16x8*)(qrow + 32 + hi * 16);
  q1.h8[1] = *(const bf16x8*)(qrow + 32 + hi * 16 + 8);

  f32x8 o0 = {}, o1 = {}, o2 = {}, o3 = {};
  float mrow[8], lrow[8];
#pragma unroll
  for (int r = 0; r < 8; ++r) { mrow[r] = -1e30f; lrow[r] = 0.0f; }

  for (int jb = 0; jb < qb + 16; jb += 32) {
    // ---- S = Q @ K^T for 32 keys (two 16-col fragments) ----
    f32x8 s0 = {}, s1 = {};
    const bf16_t* k0 = kbase + (long)(jb + rc) * D3n + hi * 16;
    const bf16_t* k1 = kbase + (long)(jb + 16 + rc) * D3n + hi * 16;
    FragB16 fb;
    fb.h8[0] = *(const bf16x8*)(k0);      fb.h8[1] = *(const bf16x8*)(k0 + 8);
    s0 = __builtin_amdgcn_wmma_f32_16x16x32_bf16(false, q0.v, false, fb.v, (short)0, s0, false, false);
    fb.h8[0] = *(const bf16x8*)(k0 + 32); fb.h8[1] = *(const bf16x8*)(k0 + 40);
    s0 = __builtin_amdgcn_wmma_f32_16x16x32_bf16(false, q1.v, false, fb.v, (short)0, s0, false, false);
    fb.h8[0] = *(const bf16x8*)(k1);      fb.h8[1] = *(const bf16x8*)(k1 + 8);
    s1 = __builtin_amdgcn_wmma_f32_16x16x32_bf16(false, q0.v, false, fb.v, (short)0, s1, false, false);
    fb.h8[0] = *(const bf16x8*)(k1 + 32); fb.h8[1] = *(const bf16x8*)(k1 + 40);
    s1 = __builtin_amdgcn_wmma_f32_16x16x32_bf16(false, q1.v, false, fb.v, (short)0, s1, false, false);

    // ---- scale, causal mask, online softmax (row stats via 16-lane xor) ----
    float scl[8], p0[8], p1[8];
#pragma unroll
    for (int r = 0; r < 8; ++r) {
      const int qg = qb + r + hi * 8;
      float v0 = (jb + rc <= qg)      ? s0[r] * kScale : -1e30f;
      float v1 = (jb + 16 + rc <= qg) ? s1[r] * kScale : -1e30f;
      float mt = fmaxf(v0, v1);
#pragma unroll
      for (int msk = 1; msk < 16; msk <<= 1) mt = fmaxf(mt, __shfl_xor(mt, msk, 32));
      const float mn = fmaxf(mrow[r], mt);
      scl[r] = __expf(mrow[r] - mn);
      p0[r] = __expf(v0 - mn);
      p1[r] = __expf(v1 - mn);
      float rs = p0[r] + p1[r];
#pragma unroll
      for (int msk = 1; msk < 16; msk <<= 1) rs += __shfl_xor(rs, msk, 32);
      lrow[r] = lrow[r] * scl[r] + rs;
      mrow[r] = mn;
    }
#pragma unroll
    for (int r = 0; r < 8; ++r) { o0[r] *= scl[r]; o1[r] *= scl[r]; o2[r] *= scl[r]; o3[r] *= scl[r]; }

    // ---- C-layout P -> LDS -> A-layout fragment (same wave; s_wait_dscnt) ----
#pragma unroll
    for (int r = 0; r < 8; ++r) {
      pshm[wave][(r + hi * 8) * 32 + rc]      = (bf16_t)p0[r];
      pshm[wave][(r + hi * 8) * 32 + 16 + rc] = (bf16_t)p1[r];
    }
    asm volatile("s_wait_dscnt 0" ::: "memory");
    FragB16 fp;
    const bf16_t* pp = &pshm[wave][rc * 32];
    fp.h8[0] = *(const bf16x8*)(pp + hi * 16);
    fp.h8[1] = *(const bf16x8*)(pp + hi * 16 + 8);

    // ---- O += P @ V (4 column tiles of HD=64) ----
    const bf16_t* vp = vbase + (long)rc * Tn + jb + hi * 16;
    FragB16 fv;
    fv.h8[0] = *(const bf16x8*)(vp);               fv.h8[1] = *(const bf16x8*)(vp + 8);
    o0 = __builtin_amdgcn_wmma_f32_16x16x32_bf16(false, fp.v, false, fv.v, (short)0, o0, false, false);
    fv.h8[0] = *(const bf16x8*)(vp + 16 * Tn);     fv.h8[1] = *(const bf16x8*)(vp + 16 * Tn + 8);
    o1 = __builtin_amdgcn_wmma_f32_16x16x32_bf16(false, fp.v, false, fv.v, (short)0, o1, false, false);
    fv.h8[0] = *(const bf16x8*)(vp + 32 * Tn);     fv.h8[1] = *(const bf16x8*)(vp + 32 * Tn + 8);
    o2 = __builtin_amdgcn_wmma_f32_16x16x32_bf16(false, fp.v, false, fv.v, (short)0, o2, false, false);
    fv.h8[0] = *(const bf16x8*)(vp + 48 * Tn);     fv.h8[1] = *(const bf16x8*)(vp + 48 * Tn + 8);
    o3 = __builtin_amdgcn_wmma_f32_16x16x32_bf16(false, fp.v, false, fv.v, (short)0, o3, false, false);
  }

  // ---- normalize and store y (bf16, [B*T, D], head-sliced columns) ----
#pragma unroll
  for (int r = 0; r < 8; ++r) {
    const float inv = 1.0f / lrow[r];
    const long yo = ((long)(bb * Tn + qb + r + hi * 8)) * Dn + hh * HDn + rc;
    y[yo]      = (bf16_t)(o0[r] * inv);
    y[yo + 16] = (bf16_t)(o1[r] * inv);
    y[yo + 32] = (bf16_t)(o2[r] * inv);
    y[yo + 48] = (bf16_t)(o3[r] * inv);
  }
}

// ---------------------------------------------------------------------------
// Host orchestration
// ---------------------------------------------------------------------------
extern "C" void kernel_launch(void* const* d_in, const int* in_sizes, int n_in,
                              void* d_out, int out_size, void* d_ws, size_t ws_size,
                              hipStream_t stream) {
  (void)in_sizes; (void)n_in; (void)out_size; (void)ws_size;
  const int*   idx     = (const int*)  d_in[0];
  const float* tok_emb = (const float*)d_in[1];
  const float* ln1_s   = (const float*)d_in[2];
  const float* ln1_b   = (const float*)d_in[3];
  const float* qkv_w   = (const float*)d_in[4];
  const float* qkv_b   = (const float*)d_in[5];
  const float* out_w   = (const float*)d_in[6];
  const float* out_b   = (const float*)d_in[7];
  const float* ln2_s   = (const float*)d_in[8];
  const float* ln2_b   = (const float*)d_in[9];
  const float* mlp_w1  = (const float*)d_in[10];
  const float* mlp_b1  = (const float*)d_in[11];
  const float* mlp_w2  = (const float*)d_in[12];
  const float* mlp_b2  = (const float*)d_in[13];
  const float* lnf_s   = (const float*)d_in[14];
  const float* lnf_b   = (const float*)d_in[15];
  const float* head_w  = (const float*)d_in[16];
  float* logits = (float*)d_out;

  char* ws = (char*)d_ws;
  size_t off = 0;
  auto take = [&](size_t bytes) -> char* {
    char* p = ws + off;
    off = (off + bytes + 255) & ~(size_t)255;
    return p;
  };
  float*  x    = (float*) take((size_t)Mn * Dn * 4);        // residual stream
  bf16_t* xb   = (bf16_t*)take((size_t)Mn * Dn * 2);        // LN output
  bf16_t* qkvb = (bf16_t*)take((size_t)Mn * D3n * 2);       // fused qkv
  bf16_t* vTb  = (bf16_t*)take((size_t)Bn * Dn * Tn * 2);   // V transposed
  bf16_t* yb   = (bf16_t*)take((size_t)Mn * Dn * 2);        // attention out
  bf16_t* h1   = (bf16_t*)take((size_t)Mn * FFn * 2);       // MLP hidden
  bf16_t* Wqkv = (bf16_t*)take((size_t)Ln * Dn * D3n * 2);
  bf16_t* Wout = (bf16_t*)take((size_t)Ln * Dn * Dn * 2);
  bf16_t* W1   = (bf16_t*)take((size_t)Ln * Dn * FFn * 2);
  bf16_t* W2   = (bf16_t*)take((size_t)Ln * FFn * Dn * 2);
  bf16_t* Whd  = (bf16_t*)take((size_t)Dn * Vn * 2);

  // 1) embeddings
  embed_kernel<<<(Mn * Dn) / 256, 256, 0, stream>>>(idx, tok_emb, x);

  // 2) weight cast + transpose to [N, K] bf16
  transpose_to_bf16<<<8192, 256, 0, stream>>>(qkv_w,  Wqkv, Dn,  D3n, Ln);
  transpose_to_bf16<<<8192, 256, 0, stream>>>(out_w,  Wout, Dn,  Dn,  Ln);
  transpose_to_bf16<<<8192, 256, 0, stream>>>(mlp_w1, W1,   Dn,  FFn, Ln);
  transpose_to_bf16<<<8192, 256, 0, stream>>>(mlp_w2, W2,   FFn, Dn,  Ln);
  transpose_to_bf16<<<8192, 256, 0, stream>>>(head_w, Whd,  Dn,  Vn,  1);

  // 64x64 tiles per wave, 8 waves per block
  constexpr int mt64 = Mn / 64;  // 32
  for (int l = 0; l < Ln; ++l) {
    // LN1 -> bf16
    layernorm_bf16<<<Mn, 256, 0, stream>>>(x, ln1_s + (size_t)l * Dn, ln1_b + (size_t)l * Dn, xb);
    // QKV projection (bf16 out, +bias)
    gemm_wmma_bf16<1, 0, 0><<<mt64 * (D3n / 64) / 8, 256, 0, stream>>>(
        xb, Wqkv + (size_t)l * D3n * Dn, qkv_b + (size_t)l * D3n, nullptr, qkvb, Mn, D3n, Dn);
    // V transpose for the P@V B-fragments
    vtranspose<<<(Bn * Dn * Tn) / 256, 256, 0, stream>>>(qkvb, vTb);
    // flash attention: B*H*(T/16) = 2048 waves = 256 blocks
    attn_flash_wmma<<<256, 256, 0, stream>>>(qkvb, vTb, yb);
    // output projection + residual (f32 into x, in place)
    gemm_wmma_bf16<0, 0, 1><<<mt64 * (Dn / 64) / 8, 256, 0, stream>>>(
        yb, Wout + (size_t)l * Dn * Dn, out_b + (size_t)l * Dn, x, x, Mn, Dn, Dn);
    // LN2 -> bf16
    layernorm_bf16<<<Mn, 256, 0, stream>>>(x, ln2_s + (size_t)l * Dn, ln2_b + (size_t)l * Dn, xb);
    // MLP up + exact GELU (bf16 out)
    gemm_wmma_bf16<1, 1, 0><<<mt64 * (FFn / 64) / 8, 256, 0, stream>>>(
        xb, W1 + (size_t)l * FFn * Dn, mlp_b1 + (size_t)l * FFn, nullptr, h1, Mn, FFn, Dn);
    // MLP down + residual (f32 into x)
    gemm_wmma_bf16<0, 0, 1><<<mt64 * (Dn / 64) / 8, 256, 0, stream>>>(
        h1, W2 + (size_t)l * Dn * FFn, mlp_b2 + (size_t)l * Dn, x, x, Mn, Dn, FFn);
  }

  // final LN + LM head (f32 logits straight to d_out)
  layernorm_bf16<<<Mn, 256, 0, stream>>>(x, lnf_s, lnf_b, xb);
  gemm_wmma_bf16<0, 0, 0><<<mt64 * (Vn / 64) / 8, 256, 0, stream>>>(
      xb, Whd, nullptr, nullptr, logits, Mn, Vn, Dn);
}
